// GroupedQueryAttention_46110768890439
// MI455X (gfx1250) — matrix operand order repaired and checked by reading, not compiled, not run on hardware
//
#include <hip/hip_runtime.h>
#include <hip/hip_bf16.h>

typedef __bf16 bf16;
typedef __attribute__((ext_vector_type(16))) __bf16 v16bf;
typedef __attribute__((ext_vector_type(8)))  float  v8f;
typedef __attribute__((ext_vector_type(8)))  unsigned int v8u;

#define B_   2
#define T_   2048
#define C_   2048
#define NH_  16
#define NKV_ 4
#define D_   128
#define NREP_ (NH_ / NKV_)
#define MROWS (B_ * T_)
#define KVC_ (NKV_ * D_)

union frag_cast { v8u u; v16bf b; };
union pack4 { float4 f; uint4 u; bf16 h[8]; };

// ---------------------------------------------------------------------------
// A-fragment (16x32 bf16, CDNA5 ISA 7.12.2): lane (m = lane&15, half = lane>>4)
// holds K-pairs. This code uses contiguous K: pair j -> K = 2j + 16*half (as in load_b_frag).
// As uint indices: {0,1,2,3}+8*half and {4,5,6,7}+8*half -> two b128 LDS loads.
// `row` points at the 32-element K-chunk for this lane's M row (4-byte aligned).
// ---------------------------------------------------------------------------
__device__ __forceinline__ v16bf load_a_frag(const bf16* row, int half) {
    const unsigned int* p = (const unsigned int*)row;
    frag_cast r;
    int o = 8 * half;
#pragma unroll
    for (int j = 0; j < 4; ++j) r.u[j] = p[j + o];
#pragma unroll
    for (int j = 0; j < 4; ++j) r.u[4 + j] = p[4 + j + o];
    return r.b;
}

// B-fragment: lane = N column; 16 contiguous K bf16 per lane-half.
// `p` is 32-byte aligned -> one v8u (2x ds_load_b128).
__device__ __forceinline__ v16bf load_b_frag(const bf16* p) {
    frag_cast r;
    r.u = *(const v8u*)p;
    return r.b;
}

// ---------------------------------------------------------------------------
// Tiled WMMA GEMM:  Out[M,N] = A[M,K] @ W[N,K]^T
// Block tile 64(M) x 128(N), BK=32. 8 waves: wave w -> M-block (w&3),
// N-strip of 64 (w>>2), 4 accumulator tiles of 16x16.
// ---------------------------------------------------------------------------
template <bool A_F32, bool OUT_F32>
__global__ void __launch_bounds__(256)
gemm_wmma_kernel(const void* __restrict__ Ap, const float* __restrict__ W,
                 void* __restrict__ Op, int M, int N, int K) {
    __shared__ alignas(32) bf16 As[64 * 32];
    __shared__ alignas(32) bf16 Bs[128 * 32];

    const int tid  = threadIdx.x;
    const int wave = tid >> 5, lane = tid & 31;
    const int half = lane >> 4, nloc = lane & 15;
    const int mw = wave & 3, nw = wave >> 2;
    const int m0 = blockIdx.y * 64, n0 = blockIdx.x * 128;

    const float* Af = (const float*)Ap;
    const bf16*  Ab = (const bf16*)Ap;

    v8f acc[4];
#pragma unroll
    for (int t = 0; t < 4; ++t)
#pragma unroll
        for (int r = 0; r < 8; ++r) acc[t][r] = 0.0f;

    for (int k0 = 0; k0 < K; k0 += 32) {
        // ---- stage A tile (64x32) as bf16 ----
        if constexpr (A_F32) {
            for (int e = tid; e < 64 * 8; e += 256) {        // float4 chunks
                int m = e >> 3, c4 = e & 7;
                pack4 pk;
                pk.f = *(const float4*)&Af[(size_t)(m0 + m) * K + k0 + c4 * 4];
                bf16 h0 = (bf16)pk.f.x, h1 = (bf16)pk.f.y;
                bf16 h2 = (bf16)pk.f.z, h3 = (bf16)pk.f.w;
                bf16* dst = &As[m * 32 + c4 * 4];
                dst[0] = h0; dst[1] = h1; dst[2] = h2; dst[3] = h3;
            }
        } else {
            for (int e = tid; e < 64 * 4; e += 256) {        // uint4 chunks (8 bf16)
                int m = e >> 2, c8 = e & 3;
                *(uint4*)&As[m * 32 + c8 * 8] =
                    *(const uint4*)&Ab[(size_t)(m0 + m) * K + k0 + c8 * 8];
            }
        }
        // ---- stage W tile (128x32) as bf16 ----
        for (int e = tid; e < 128 * 8; e += 256) {           // float4 chunks
            int n = e >> 3, c4 = e & 7;
            pack4 pk;
            pk.f = *(const float4*)&W[(size_t)(n0 + n) * K + k0 + c4 * 4];
            bf16* dst = &Bs[n * 32 + c4 * 4];
            dst[0] = (bf16)pk.f.x; dst[1] = (bf16)pk.f.y;
            dst[2] = (bf16)pk.f.z; dst[3] = (bf16)pk.f.w;
        }
        // Prefetch next W K-tile into cache (global_prefetch_b8).
        if (k0 + 32 < K)
            __builtin_prefetch(&W[(size_t)(n0 + (tid >> 1)) * K + k0 + 32 + (tid & 1) * 16], 0, 1);
        __syncthreads();

        v16bf a = load_a_frag(&As[(mw * 16 + nloc) * 32], half);
#pragma unroll
        for (int t = 0; t < 4; ++t) {
            v16bf b = load_b_frag(&Bs[(nw * 64 + t * 16 + nloc) * 32 + half * 16]);
            acc[t] = __builtin_amdgcn_wmma_f32_16x16x32_bf16(
                false, a, false, b, (short)0, acc[t], false, false);
        }
        __syncthreads();
    }

    // C/D layout: VGPR r, lane-half -> row r + 8*half; lane&15 -> col.
#pragma unroll
    for (int t = 0; t < 4; ++t)
#pragma unroll
        for (int r = 0; r < 8; ++r) {
            size_t row = (size_t)(m0 + mw * 16 + r + 8 * half);
            size_t col = (size_t)(n0 + nw * 64 + t * 16 + nloc);
            if constexpr (OUT_F32)
                ((float*)Op)[row * N + col] = acc[t][r];
            else
                ((bf16*)Op)[row * N + col] = (bf16)acc[t][r];
        }
}

// ---------------------------------------------------------------------------
// RoPE on a [MROWS, nheads*128] bf16 tensor (one thread per even/odd pair).
// ---------------------------------------------------------------------------
__global__ void __launch_bounds__(256)
rope_kernel(bf16* __restrict__ t, int nheads) {
    int idx = blockIdx.x * 256 + threadIdx.x;
    int per = nheads * 64;
    int row = idx / per;
    int rem = idx - row * per;
    int h = rem >> 6, p = rem & 63;
    int pos = row & (T_ - 1);

    float inv = __powf(10000.0f, -(2.0f * (float)p) / 128.0f);
    float ang = (float)pos * inv;
    float c = __cosf(ang), s = __sinf(ang);

    size_t base = (size_t)row * (nheads * 128) + h * 128 + 2 * p;
    float te = (float)t[base];
    float to = (float)t[base + 1];
    t[base]     = (bf16)(te * c - to * s);
    t[base + 1] = (bf16)(te * s + to * c);
}

// ---------------------------------------------------------------------------
// Causal flash attention. Block = (b, head, 64 query rows), 8 waves.
// Q tile 64x128 resident in LDS; stream 32-key K/V tiles. V staged TRANSPOSED
// ([d][key]) so P@V B-fragments are contiguous 32-byte LDS loads.
// ---------------------------------------------------------------------------
__global__ void __launch_bounds__(256)
fa_kernel(const bf16* __restrict__ q, const bf16* __restrict__ k,
          const bf16* __restrict__ v, bf16* __restrict__ o) {
    __shared__ alignas(32) bf16  Qs[64 * 128];   // [qrow][d]
    __shared__ alignas(32) bf16  Ks[32 * 128];   // [key][d]
    __shared__ alignas(32) bf16  Vs[128 * 32];   // [d][key]  (transposed)
    __shared__ alignas(32) float Ss[64 * 32];
    __shared__ alignas(32) bf16  Ps[64 * 32];
    __shared__ float m_s[64], l_s[64], al_s[64];

    const int tid  = threadIdx.x;
    const int wave = tid >> 5, lane = tid & 31;
    const int half = lane >> 4, nloc = lane & 15;
    const int mblk = wave & 3, nquad = wave >> 2;
    const int bh = blockIdx.x;
    const int b = bh / NH_, h = bh % NH_, hkv = h / NREP_;
    const int q0 = blockIdx.y * 64;
    const float scale = 0.08838834764831845f;    // 1/sqrt(128)

    // Stage Q tile: 64x128 bf16 as uint4 (8 bf16) chunks.
    for (int e = tid; e < 64 * 16; e += 256) {
        int r = e >> 4, c8 = e & 15;
        *(uint4*)&Qs[r * 128 + c8 * 8] =
            *(const uint4*)&q[(size_t)(b * T_ + q0 + r) * C_ + h * D_ + c8 * 8];
    }
    if (tid < 64) { m_s[tid] = -1e30f; l_s[tid] = 0.0f; }

    v8f acc[4];
#pragma unroll
    for (int t = 0; t < 4; ++t)
#pragma unroll
        for (int r = 0; r < 8; ++r) acc[t][r] = 0.0f;
    __syncthreads();

    const int jend = q0 + 63;
    for (int j0 = 0; j0 <= jend; j0 += 32) {
        // ---- stage K (row-major) and V (transposed) ----
        for (int e = tid; e < 32 * 16; e += 256) {
            int r = e >> 4, c8 = e & 15;
            size_t gi = (size_t)(b * T_ + j0 + r) * KVC_ + hkv * D_ + c8 * 8;
            *(uint4*)&Ks[r * 128 + c8 * 8] = *(const uint4*)&k[gi];
            pack4 pv;
            pv.u = *(const uint4*)&v[gi];
#pragma unroll
            for (int i = 0; i < 8; ++i) Vs[(c8 * 8 + i) * 32 + r] = pv.h[i];
        }
        __syncthreads();

        // ---- S = Q @ K^T (this wave's 16x16 tile, 4 WMMAs over D) ----
        v8f s;
#pragma unroll
        for (int r = 0; r < 8; ++r) s[r] = 0.0f;
#pragma unroll
        for (int kc = 0; kc < 4; ++kc) {
            v16bf a  = load_a_frag(&Qs[(mblk * 16 + nloc) * 128 + kc * 32], half);
            v16bf bb = load_b_frag(&Ks[(nquad * 16 + nloc) * 128 + kc * 32 + half * 16]);
            s = __builtin_amdgcn_wmma_f32_16x16x32_bf16(
                false, a, false, bb, (short)0, s, false, false);
        }
#pragma unroll
        for (int r = 0; r < 8; ++r)
            Ss[(mblk * 16 + r + 8 * half) * 32 + nquad * 16 + nloc] = s[r] * scale;
        __syncthreads();

        // ---- online softmax row pass (64 threads, one per query row) ----
        if (tid < 64) {
            int gi = q0 + tid;
            float mold = m_s[tid];
            float mx = mold;
            float rv[32];
#pragma unroll
            for (int c = 0; c < 32; ++c) {
                float xv = (j0 + c <= gi) ? Ss[tid * 32 + c] : -1e30f;
                rv[c] = xv;
                mx = fmaxf(mx, xv);
            }
            float alpha = __expf(mold - mx);
            float lsum  = l_s[tid] * alpha;
#pragma unroll
            for (int c = 0; c < 32; ++c) {
                float p = (rv[c] > -1e29f) ? __expf(rv[c] - mx) : 0.0f;
                Ps[tid * 32 + c] = (bf16)p;
                lsum += p;
            }
            m_s[tid] = mx;
            l_s[tid] = lsum;
            al_s[tid] = alpha;
        }
        __syncthreads();

        // ---- O = O*alpha + P @ V (single K=32 WMMA per tile) ----
        float arow[8];
#pragma unroll
        for (int r = 0; r < 8; ++r) arow[r] = al_s[mblk * 16 + r + 8 * half];

        v16bf pa = load_a_frag(&Ps[(mblk * 16 + nloc) * 32], half);
#pragma unroll
        for (int t = 0; t < 4; ++t) {
            int dcol = nquad * 64 + t * 16 + nloc;
            v16bf vbf = load_b_frag(&Vs[dcol * 32 + half * 16]);
#pragma unroll
            for (int r = 0; r < 8; ++r) acc[t][r] *= arow[r];
            acc[t] = __builtin_amdgcn_wmma_f32_16x16x32_bf16(
                false, pa, false, vbf, (short)0, acc[t], false, false);
        }
        __syncthreads();
    }

    // ---- normalize by l and store [B*T, NH*D] bf16 ----
    float linv[8];
#pragma unroll
    for (int r = 0; r < 8; ++r) {
        float l = l_s[mblk * 16 + r + 8 * half];
        linv[r] = (l > 0.0f) ? 1.0f / l : 0.0f;
    }
#pragma unroll
    for (int t = 0; t < 4; ++t)
#pragma unroll
        for (int r = 0; r < 8; ++r) {
            size_t row = (size_t)(b * T_ + q0 + mblk * 16 + r + 8 * half);
            size_t col = (size_t)(h * D_ + nquad * 64 + t * 16 + nloc);
            o[row * C_ + col] = (bf16)(acc[t][r] * linv[r]);
        }
}

// ---------------------------------------------------------------------------
extern "C" void kernel_launch(void* const* d_in, const int* in_sizes, int n_in,
                              void* d_out, int out_size, void* d_ws, size_t ws_size,
                              hipStream_t stream) {
    (void)in_sizes; (void)n_in; (void)out_size; (void)ws_size;
    const float* x  = (const float*)d_in[0];
    const float* Wq = (const float*)d_in[1];
    const float* Wk = (const float*)d_in[2];
    const float* Wv = (const float*)d_in[3];
    const float* Wo = (const float*)d_in[4];
    float* out = (float*)d_out;

    bf16* qb = (bf16*)d_ws;                               // [MROWS, C_]
    bf16* kb = qb + (size_t)MROWS * C_;                   // [MROWS, KVC_]
    bf16* vb = kb + (size_t)MROWS * KVC_;                 // [MROWS, KVC_]
    bf16* ab = vb + (size_t)MROWS * KVC_;                 // [MROWS, C_]

    dim3 blk(256);

    // Projections (f32 in -> bf16 out)
    gemm_wmma_kernel<true, false>
        <<<dim3(C_ / 128, MROWS / 64), blk, 0, stream>>>(x, Wq, qb, MROWS, C_, C_);
    gemm_wmma_kernel<true, false>
        <<<dim3(KVC_ / 128, MROWS / 64), blk, 0, stream>>>(x, Wk, kb, MROWS, KVC_, C_);
    gemm_wmma_kernel<true, false>
        <<<dim3(KVC_ / 128, MROWS / 64), blk, 0, stream>>>(x, Wv, vb, MROWS, KVC_, C_);

    // RoPE on q and k
    rope_kernel<<<(MROWS * NH_ * 64) / 256, blk, 0, stream>>>(qb, NH_);
    rope_kernel<<<(MROWS * NKV_ * 64) / 256, blk, 0, stream>>>(kb, NKV_);

    // Causal flash attention (GQA: head h uses kv head h/NREP_)
    fa_kernel<<<dim3(B_ * NH_, T_ / 64), blk, 0, stream>>>(qb, kb, vb, ab);

    // Output projection (bf16 in -> f32 out)
    gemm_wmma_kernel<false, true>
        <<<dim3(C_ / 128, MROWS / 64), blk, 0, stream>>>(ab, Wo, out, MROWS, C_, C_);
}
